// FastAttention_5377299054789
// MI455X (gfx1250) — matrix operand order repaired and checked by reading, not compiled, not run on hardware
//
#include <hip/hip_runtime.h>
#include <hip/hip_bf16.h>
#include <stdint.h>

#define B_  4
#define L_  4096
#define E_  1024
#define H_  16
#define D_  64
#define KM_ 64
#define BL_ (B_*L_)
#define N3_ (3*E_)

typedef __bf16 bf16;
typedef __attribute__((ext_vector_type(16))) __bf16  v16bf;
typedef __attribute__((ext_vector_type(8)))  float   v8f;
typedef __attribute__((ext_vector_type(4)))  unsigned uint4v;
typedef __attribute__((ext_vector_type(2)))  unsigned uint2v;
typedef __attribute__((ext_vector_type(4)))  float   float4v;

union PK4  { bf16 h[4];  uint2v u; };
union PK8  { bf16 h[8];  uint4v u; };
union PK16 { bf16 h[16]; uint4v u[2]; };
struct FragBF { union { uint4v u[2]; v16bf v; }; };

// A fragment (16x32 bf16, row-major source, row stride ld elements):
// lanes 0-15 hold rows M=0..15 with K=0..7 (V0-3) and K=16..23 (V4-7);
// lanes 16-31 hold same rows with K=8..15 / K=24..31.
__device__ __forceinline__ FragBF ldfragA(const bf16* p, int ld, int lane) {
  FragBF f;
  const bf16* q = p + (lane & 15) * ld + (lane >> 4) * 8;
  f.u[0] = *(const uint4v*)(q);
  f.u[1] = *(const uint4v*)(q + 16);
  return f;
}
// B fragment (32x16 bf16) from Bt (row-major [N][K]): lane&15 = column N,
// lanes 0-15 hold K=0..7 / K=16..23, lanes 16-31 hold K=8..15 / K=24..31 (same as A).
__device__ __forceinline__ FragBF ldfragB(const bf16* p, int ld, int lane) {
  FragBF f;
  const bf16* q = p + (lane & 15) * ld + (lane >> 4) * 8;
  f.u[0] = *(const uint4v*)(q);
  f.u[1] = *(const uint4v*)(q + 16);
  return f;
}
__device__ __forceinline__ v8f wmma_bf16(const FragBF& a, const FragBF& b, v8f c) {
  return __builtin_amdgcn_wmma_f32_16x16x32_bf16(false, a.v, false, b.v, (short)0, c, false, false);
}

// CDNA5 async data mover: copy 16 B per lane, global -> LDS, tracked by ASYNCcnt.
// LDS operand = wave-relative byte offset (= low 32 bits of the generic shared ptr).
__device__ __forceinline__ void async_cp16(void* lds, const void* g) {
  unsigned loff = (unsigned)(uintptr_t)lds;
  unsigned long long ga = (unsigned long long)(uintptr_t)g;
  asm volatile("global_load_async_to_lds_b128 %0, %1, off"
               :: "v"(loff), "v"(ga) : "memory");
}
#define WAIT_ASYNC(n) asm volatile("s_wait_asynccnt %0" :: "n"(n) : "memory")

// ---------------- f32 -> bf16 elementwise ----------------
__global__ void k_cvt_bf16(const float* __restrict__ in, bf16* __restrict__ out, int n4) {
  int i = blockIdx.x * 256 + threadIdx.x;
  if (i >= n4) return;
  float4v v = *(const float4v*)(in + (size_t)i * 4);
  PK4 o;
  o.h[0] = (bf16)v.x; o.h[1] = (bf16)v.y; o.h[2] = (bf16)v.z; o.h[3] = (bf16)v.w;
  *(uint2v*)(out + (size_t)i * 4) = o.u;
}

// ---------------- f32 [R][C] -> bf16 [C][R] transpose ----------------
__global__ void k_transpose_cvt(const float* __restrict__ in, bf16* __restrict__ out,
                                int R, int C, long inStride, long outStride) {
  __shared__ float ts[64][65];
  const float* inp = in + (size_t)blockIdx.z * inStride;
  bf16* outp = out + (size_t)blockIdx.z * outStride;
  int t = threadIdx.x;
  int r0 = blockIdx.y * 64, c0 = blockIdx.x * 64;
#pragma unroll
  for (int j = 0; j < 4; ++j) {
    int ch = t + 256 * j;
    int row = ch >> 4, cc = (ch & 15) * 4;
    float4v v = *(const float4v*)(inp + (size_t)(r0 + row) * C + c0 + cc);
    ts[row][cc + 0] = v.x; ts[row][cc + 1] = v.y;
    ts[row][cc + 2] = v.z; ts[row][cc + 3] = v.w;
  }
  __syncthreads();
#pragma unroll
  for (int j = 0; j < 2; ++j) {
    int ch = t + 256 * j;
    int orow = ch >> 3, oc = (ch & 7) * 8;
    PK8 p;
#pragma unroll
    for (int ii = 0; ii < 8; ++ii) p.h[ii] = (bf16)ts[oc + ii][orow];
    *(uint4v*)(outp + (size_t)(c0 + orow) * R + r0 + oc) = p.u;
  }
}

// ---------------- big bf16 WMMA GEMM: C[M][N] = A[M][K] * Bt[N][K]^T + bias ----------------
// Double-buffered LDS tiles filled by async global->LDS copies (ASYNCcnt pipelined).
template<int OUT_BF16>
__global__ void k_gemm(const bf16* __restrict__ A, const bf16* __restrict__ Bt,
                       const float* __restrict__ bias, void* __restrict__ Cout,
                       int N, int K) {
  __shared__ __align__(16) bf16 As[2][128 * 40];
  __shared__ __align__(16) bf16 Bs[2][128 * 40];
  int t = threadIdx.x, lane = t & 31, w = t >> 5;
  int wm = w >> 2, wn = w & 3;                 // 2x4 wave grid, wave tile 64x32
  size_t m0 = (size_t)blockIdx.y * 128;
  int n0 = blockIdx.x * 128;
  int row0 = t >> 2,         col0 = (t & 3) * 8;
  int row1 = (t + 256) >> 2, col1 = ((t + 256) & 3) * 8;
  v8f acc[4][2] = {};

  auto issue = [&](int kk, int b) {            // 4 async b128 copies per thread
    async_cp16(&As[b][row0 * 40 + col0], A  + (m0 + row0) * K + kk + col0);
    async_cp16(&As[b][row1 * 40 + col1], A  + (m0 + row1) * K + kk + col1);
    async_cp16(&Bs[b][row0 * 40 + col0], Bt + (size_t)(n0 + row0) * K + kk + col0);
    async_cp16(&Bs[b][row1 * 40 + col1], Bt + (size_t)(n0 + row1) * K + kk + col1);
  };

  issue(0, 0);
  int nk = K >> 5;
  for (int i = 0; i < nk; ++i) {
    int cur = i & 1;
    if (i + 1 < nk) {
      issue((i + 1) << 5, cur ^ 1);
      WAIT_ASYNC(4);   // in-order: <=4 outstanding => current tile's copies landed
    } else {
      WAIT_ASYNC(0);
    }
    __syncthreads();
    const bf16* as = As[cur];
    const bf16* bs = Bs[cur];
    FragBF af[4], bff[2];
#pragma unroll
    for (int mt = 0; mt < 4; ++mt)
      af[mt] = ldfragA(as + (wm * 64 + mt * 16) * 40, 40, lane);
#pragma unroll
    for (int nt = 0; nt < 2; ++nt)
      bff[nt] = ldfragB(bs + (wn * 32 + nt * 16) * 40, 40, lane);
#pragma unroll
    for (int mt = 0; mt < 4; ++mt)
#pragma unroll
      for (int nt = 0; nt < 2; ++nt)
        acc[mt][nt] = wmma_bf16(af[mt], bff[nt], acc[mt][nt]);
    __syncthreads();   // all waves done reading buf[cur] before it is refilled
  }
#pragma unroll
  for (int mt = 0; mt < 4; ++mt) {
    size_t rowb = m0 + wm * 64 + mt * 16 + 8 * (lane >> 4);
#pragma unroll
    for (int nt = 0; nt < 2; ++nt) {
      int col = n0 + wn * 32 + nt * 16 + (lane & 15);
      float bv = bias[col];
#pragma unroll
      for (int j = 0; j < 8; ++j) {
        float v = acc[mt][nt][j] + bv;
        if (OUT_BF16) ((bf16*)Cout)[(rowb + j) * N + col] = (bf16)v;
        else          ((float*)Cout)[(rowb + j) * N + col] = v;
      }
    }
  }
}

// ---------------- RoPE on q/k (in-place) + masked transpose of v -> vT[B,H,D,L] ----------------
__global__ void k_rope_vt(bf16* __restrict__ qkv, const float* __restrict__ fc,
                          const float* __restrict__ fs, const float* __restrict__ mask,
                          bf16* __restrict__ vT) {
  __shared__ float fcs[64][32];
  __shared__ float fss[64][32];
  __shared__ __align__(16) bf16 vs[64][72];
  int t = threadIdx.x;
  int b = blockIdx.y;
  int l0 = blockIdx.x * 64;
#pragma unroll
  for (int j = 0; j < 8; ++j) {
    int c = t + 256 * j; int row = c >> 5, i = c & 31;
    fcs[row][i] = fc[(l0 + row) * 32 + i];
    fss[row][i] = fs[(l0 + row) * 32 + i];
  }
  __syncthreads();
  for (int it = 0; it < 256; ++it) {
    int idx = it * 256 + t;
    int row = idx >> 10, p = idx & 1023;
    int sec = p >> 9, pin = p & 511;
    int i = pin & 31;
    int col = sec * 1024 + pin * 2;
    float c = fcs[row][i], s = fss[row][i];
    bf16* addr = qkv + (size_t)(b * L_ + l0 + row) * N3_ + col;
    union { unsigned u; bf16 h[2]; } dv;
    dv.u = *(unsigned*)addr;
    float xr = (float)dv.h[0], xi = (float)dv.h[1];
    dv.h[0] = (bf16)(xr * c - xi * s);
    dv.h[1] = (bf16)(xr * s + xi * c);
    *(unsigned*)addr = dv.u;
  }
  for (int h = 0; h < H_; ++h) {
    __syncthreads();
#pragma unroll
    for (int j = 0; j < 2; ++j) {
      int c = t + 256 * j; int row = c >> 3, c8 = (c & 7) * 8;
      int l = l0 + row;
      float mv = mask[b * L_ + l];
      PK8 pk;
      pk.u = *(const uint4v*)(qkv + (size_t)(b * L_ + l) * N3_ + 2048 + h * 64 + c8);
#pragma unroll
      for (int ii = 0; ii < 8; ++ii) pk.h[ii] = (bf16)((float)pk.h[ii] * mv);
      *(uint4v*)&vs[row][c8] = pk.u;
    }
    __syncthreads();
    int d = t >> 2, lofs = (t & 3) * 16;
    PK16 pk;
#pragma unroll
    for (int ii = 0; ii < 16; ++ii) pk.h[ii] = vs[lofs + ii][d];
    bf16* dst = vT + ((size_t)(b * H_ + h) * 64 + d) * L_ + l0 + lofs;
    *(uint4v*)dst = pk.u[0];
    *(uint4v*)(dst + 8) = pk.u[1];
  }
}

// ---------------- per-head projection: qp = relu(q @ proj), kpT = (relu(k @ proj)*mask)^T ----------------
__global__ void k_proj(const bf16* __restrict__ qkv, const bf16* __restrict__ projT,
                       const float* __restrict__ mask,
                       bf16* __restrict__ qp, bf16* __restrict__ kpT) {
  __shared__ __align__(16) bf16 As[256 * 72];
  __shared__ __align__(16) bf16 Bs[64 * 72];
  int t = threadIdx.x, lane = t & 31, w = t >> 5;
  int bh = blockIdx.y, b = bh >> 4, h = bh & 15;
  int isK = blockIdx.z;
  int l0 = blockIdx.x * 256;
  const bf16* Abase = qkv + (size_t)(b * L_ + l0) * N3_ + isK * 1024 + h * 64;
#pragma unroll
  for (int j = 0; j < 8; ++j) {
    int c = t + 256 * j; int row = c >> 3, c8 = (c & 7) * 8;
    async_cp16(As + row * 72 + c8, Abase + (size_t)row * N3_ + c8);
  }
#pragma unroll
  for (int j = 0; j < 2; ++j) {
    int c = t + 256 * j; int row = c >> 3, c8 = (c & 7) * 8;
    async_cp16(Bs + row * 72 + c8, projT + h * 4096 + row * 64 + c8);
  }
  WAIT_ASYNC(0);
  __syncthreads();
  v8f acc[2][4] = {};
#pragma unroll
  for (int kk = 0; kk < 64; kk += 32) {
    FragBF af[2], bff[4];
#pragma unroll
    for (int mt = 0; mt < 2; ++mt)
      af[mt] = ldfragA(As + (w * 32 + mt * 16) * 72 + kk, 72, lane);
#pragma unroll
    for (int nt = 0; nt < 4; ++nt)
      bff[nt] = ldfragB(Bs + (nt * 16) * 72 + kk, 72, lane);
#pragma unroll
    for (int mt = 0; mt < 2; ++mt)
#pragma unroll
      for (int nt = 0; nt < 4; ++nt)
        acc[mt][nt] = wmma_bf16(af[mt], bff[nt], acc[mt][nt]);
  }
#pragma unroll
  for (int mt = 0; mt < 2; ++mt) {
    int rowl = l0 + w * 32 + mt * 16 + 8 * (lane >> 4);
    float mrow[8];
    if (isK) {
#pragma unroll
      for (int j = 0; j < 8; ++j) mrow[j] = mask[b * L_ + rowl + j];
    }
#pragma unroll
    for (int nt = 0; nt < 4; ++nt) {
      int col = nt * 16 + (lane & 15);
      if (!isK) {
#pragma unroll
        for (int j = 0; j < 8; ++j) {
          float v = acc[mt][nt][j]; v = v > 0.f ? v : 0.f;
          qp[((size_t)bh * L_ + rowl + j) * 64 + col] = (bf16)v;
        }
      } else {
        PK8 pk;
#pragma unroll
        for (int j = 0; j < 8; ++j) {
          float v = acc[mt][nt][j]; v = v > 0.f ? v : 0.f;
          pk.h[j] = (bf16)(v * mrow[j]);
        }
        *(uint4v*)(kpT + ((size_t)bh * 64 + col) * L_ + rowl) = pk.u;
      }
    }
  }
}

// ---------------- kv[m][d] = sum_l kpT[m][l]*vT[d][l]  (per (b,h), K split across waves) ----------------
__global__ void k_kv(const bf16* __restrict__ kpT, const bf16* __restrict__ vT,
                     bf16* __restrict__ kvT, float* __restrict__ ksum) {
  __shared__ float kvpart[8][16 * 64];
  __shared__ float kspart[8][16];
  int t = threadIdx.x, lane = t & 31, w = t >> 5;
  int bh = blockIdx.x;
  int mh = w >> 1, ks = w & 1;   // 4-way m split (16 rows), 2-way K split (2048 each)
  const bf16* A  = kpT + ((size_t)bh * 64 + mh * 16) * L_;
  const bf16* Bt = vT + (size_t)bh * 64 * L_;
  v8f acc[4] = {};
  float ksa = 0.f;
  for (int k0 = ks * 2048; k0 < ks * 2048 + 2048; k0 += 32) {
    FragBF af = ldfragA(A + k0, L_, lane);
#pragma unroll
    for (int ii = 0; ii < 16; ++ii) ksa += (float)af.v[ii];
    FragBF bff[4];
#pragma unroll
    for (int nt = 0; nt < 4; ++nt)
      bff[nt] = ldfragB(Bt + (size_t)(nt * 16) * L_ + k0, L_, lane);
#pragma unroll
    for (int nt = 0; nt < 4; ++nt)
      acc[nt] = wmma_bf16(af, bff[nt], acc[nt]);
  }
#pragma unroll
  for (int nt = 0; nt < 4; ++nt)
#pragma unroll
    for (int j = 0; j < 8; ++j) {
      int m15 = 8 * (lane >> 4) + j;
      int d = nt * 16 + (lane & 15);
      kvpart[w][m15 * 64 + d] = acc[nt][j];
    }
  {
    float pairs = ksa + __shfl_xor(ksa, 16, 32);
    if (lane < 16) kspart[w][lane] = pairs;
  }
  __syncthreads();
  {
    int d = t >> 2, mofs = (t & 3) * 16;
    PK16 pk;
#pragma unroll
    for (int ii = 0; ii < 16; ++ii) {
      int m = mofs + ii;
      int wb = (m >> 4) * 2, m15 = m & 15;
      float v = kvpart[wb][m15 * 64 + d] + kvpart[wb + 1][m15 * 64 + d];
      pk.h[ii] = (bf16)v;
    }
    bf16* dst = kvT + (size_t)bh * 4096 + d * 64 + mofs;  // kvT[d][m]
    *(uint4v*)dst = pk.u[0];
    *(uint4v*)(dst + 8) = pk.u[1];
  }
  if (t < 64) {
    int wb = (t >> 4) * 2, m15 = t & 15;
    ksum[bh * 64 + t] = kspart[wb][m15] + kspart[wb + 1][m15];
  }
}

// ---------------- norm[b,h,l] = dot(qp[b,h,l,:], ksum[b,h,:]) ----------------
__global__ void k_norm(const bf16* __restrict__ qp, const float* __restrict__ ksum,
                       float* __restrict__ nrm) {
  __shared__ float ks[64];
  int t = threadIdx.x;
  size_t idx0 = (size_t)blockIdx.x * 256;
  int bh = (int)(idx0 >> 12);
  if (t < 64) ks[t] = ksum[bh * 64 + t];
  __syncthreads();
  size_t idx = idx0 + t;
  const bf16* row = qp + idx * 64;
  float acc = 0.f;
#pragma unroll
  for (int j = 0; j < 8; ++j) {
    PK8 pk; pk.u = *(const uint4v*)(row + j * 8);
#pragma unroll
    for (int ii = 0; ii < 8; ++ii) acc += (float)pk.h[ii] * ks[j * 8 + ii];
  }
  nrm[idx] = acc;
}

// ---------------- ctx = qp @ kv, /(norm+1e-4), write y[B,L,E] bf16 ----------------
__global__ void k_ctx(const bf16* __restrict__ qp, const bf16* __restrict__ kvT,
                      const float* __restrict__ nrm, bf16* __restrict__ y) {
  __shared__ __align__(16) bf16 As[256 * 72];
  __shared__ __align__(16) bf16 Bs[64 * 72];
  int t = threadIdx.x, lane = t & 31, w = t >> 5;
  int bh = blockIdx.y, b = bh >> 4, h = bh & 15;
  int l0 = blockIdx.x * 256;
  const bf16* Abase = qp + ((size_t)bh * L_ + l0) * 64;
#pragma unroll
  for (int j = 0; j < 8; ++j) {
    int c = t + 256 * j; int row = c >> 3, c8 = (c & 7) * 8;
    async_cp16(As + row * 72 + c8, Abase + (size_t)row * 64 + c8);
  }
#pragma unroll
  for (int j = 0; j < 2; ++j) {
    int c = t + 256 * j; int row = c >> 3, c8 = (c & 7) * 8;
    async_cp16(Bs + row * 72 + c8, kvT + (size_t)bh * 4096 + row * 64 + c8);
  }
  WAIT_ASYNC(0);
  __syncthreads();
  v8f acc[2][4] = {};
#pragma unroll
  for (int kk = 0; kk < 64; kk += 32) {
    FragBF af[2], bff[4];
#pragma unroll
    for (int mt = 0; mt < 2; ++mt)
      af[mt] = ldfragA(As + (w * 32 + mt * 16) * 72 + kk, 72, lane);
#pragma unroll
    for (int nt = 0; nt < 4; ++nt)
      bff[nt] = ldfragB(Bs + (nt * 16) * 72 + kk, 72, lane);
#pragma unroll
    for (int mt = 0; mt < 2; ++mt)
#pragma unroll
      for (int nt = 0; nt < 4; ++nt)
        acc[mt][nt] = wmma_bf16(af[mt], bff[nt], acc[mt][nt]);
  }
#pragma unroll
  for (int mt = 0; mt < 2; ++mt) {
    int rowl = l0 + w * 32 + mt * 16 + 8 * (lane >> 4);
    float inv[8];
#pragma unroll
    for (int j = 0; j < 8; ++j) inv[j] = 1.f / (nrm[(size_t)bh * L_ + rowl + j] + 1e-4f);
#pragma unroll
    for (int nt = 0; nt < 4; ++nt) {
      int col = nt * 16 + (lane & 15);
#pragma unroll
      for (int j = 0; j < 8; ++j) {
        float v = acc[mt][nt][j] * inv[j];
        y[((size_t)(b * L_ + rowl + j)) * E_ + h * 64 + col] = (bf16)v;
      }
    }
  }
}

extern "C" void kernel_launch(void* const* d_in, const int* in_sizes, int n_in,
                              void* d_out, int out_size, void* d_ws, size_t ws_size,
                              hipStream_t stream) {
  const float* x    = (const float*)d_in[0];
  const float* fc   = (const float*)d_in[1];
  const float* fs   = (const float*)d_in[2];
  const float* mask = (const float*)d_in[3];
  const float* Wqkv = (const float*)d_in[4];
  const float* bqkv = (const float*)d_in[5];
  const float* proj = (const float*)d_in[6];
  const float* Wout = (const float*)d_in[7];
  const float* bout = (const float*)d_in[8];
  (void)in_sizes; (void)n_in; (void)out_size; (void)ws_size;

  char* ws = (char*)d_ws;
  bf16*  xbf   = (bf16*)(ws + 0);           // 33,554,432
  bf16*  WqkvT = (bf16*)(ws + 33554432);    //  6,291,456
  bf16*  WoutT = (bf16*)(ws + 39845888);    //  2,097,152
  bf16*  projT = (bf16*)(ws + 41943040);    //    131,072
  bf16*  qkv   = (bf16*)(ws + 42074112);    // 100,663,296
  bf16*  vT    = (bf16*)(ws + 142737408);   // 33,554,432
  bf16*  qp    = (bf16*)(ws + 176291840);   // 33,554,432
  bf16*  kpT   = (bf16*)(ws + 209846272);   // 33,554,432
  bf16*  kvT   = (bf16*)(ws + 243400704);   //    524,288
  float* ksum  = (float*)(ws + 243924992);  //     16,384
  float* nrm   = (float*)(ws + 243941376);  //  1,048,576
  bf16*  y     = (bf16*)(ws + 244989952);   // 33,554,432  (total ~278.5 MB)

  k_cvt_bf16<<<dim3((BL_ * E_ / 4) / 256), 256, 0, stream>>>(x, xbf, BL_ * E_ / 4);
  k_transpose_cvt<<<dim3(48, 16, 1), 256, 0, stream>>>(Wqkv, WqkvT, 1024, 3072, 0, 0);
  k_transpose_cvt<<<dim3(16, 16, 1), 256, 0, stream>>>(Wout, WoutT, 1024, 1024, 0, 0);
  k_transpose_cvt<<<dim3(1, 1, 16), 256, 0, stream>>>(proj, projT, 64, 64, 4096, 4096);
  k_gemm<1><<<dim3(24, 128), 256, 0, stream>>>(xbf, WqkvT, bqkv, qkv, 3072, 1024);
  k_rope_vt<<<dim3(64, 4), 256, 0, stream>>>(qkv, fc, fs, mask, vT);
  k_proj<<<dim3(16, 64, 2), 256, 0, stream>>>(qkv, projT, mask, qp, kpT);
  k_kv<<<dim3(64), 256, 0, stream>>>(kpT, vT, kvT, ksum);
  k_norm<<<dim3(1024), 256, 0, stream>>>(qp, ksum, nrm);
  k_ctx<<<dim3(16, 64), 256, 0, stream>>>(qp, kvT, nrm, y);
  k_gemm<0><<<dim3(8, 128), 256, 0, stream>>>(y, WoutT, bout, d_out, 1024, 1024);
}
